// BiRNN_50517405336211
// MI455X (gfx1250) — matrix operand order repaired and checked by reading, not compiled, not run on hardware
//
#include <hip/hip_runtime.h>

// ---------------------------------------------------------------------------
// Types for CDNA5 WMMA (wave32): D(f32 16x16) = A(bf16 16x32) * B(bf16 32x16) + C
// ---------------------------------------------------------------------------
typedef __attribute__((ext_vector_type(16))) __bf16 v16bf;
typedef __attribute__((ext_vector_type(8)))  __bf16 bf16x8;
typedef __attribute__((ext_vector_type(8)))  float  v8f;

union Frag {
    v16bf  v;
    bf16x8 h[2];
};

__device__ __forceinline__ v8f wmma_bf16(const Frag& a, const Frag& b, v8f c) {
    return __builtin_amdgcn_wmma_f32_16x16x32_bf16(
        /*neg_a=*/false, a.v, /*neg_b=*/false, b.v,
        /*c_mod=*/(short)0, c, /*reuse_a=*/false, /*reuse_b=*/false);
}

__device__ __forceinline__ float sigmoid_f(float x) {
    return 1.0f / (1.0f + __expf(-x));
}
__device__ __forceinline__ float tanh_f(float x) {
    // tanh(x) = 1 - 2/(exp(2x)+1); saturates correctly at +/-1
    return 1.0f - 2.0f / (__expf(2.0f * x) + 1.0f);
}

static constexpr int Bn = 64, Tn = 512, In = 256, Hn = 256;
static constexpr int GN = 3 * Hn;        // 768 gate columns per direction
static constexpr int NC = 2 * GN;        // 1536 combined columns (fwd|bwd)

// ---------------------------------------------------------------------------
// fp32 -> bf16 conversion (RNE via compiler cvt)
// ---------------------------------------------------------------------------
__global__ void cvt_bf16_kernel(const float* __restrict__ s, __bf16* __restrict__ d, int n) {
    int i = blockIdx.x * blockDim.x + threadIdx.x;
    if (i < n) d[i] = (__bf16)s[i];
}

// ---------------------------------------------------------------------------
// gi GEMM: G[M=32768, N=1536] = Xb[M,256] * Wic^T + bias
//   row m = b*T + t (x is [B,T,I] row-major); Wic = [Wi_f ; Wi_b] as [1536,256]
//   Each wave computes a 16(M) x 64(N) strip: 4 accumulators, A reused 4x.
// ---------------------------------------------------------------------------
__global__ __launch_bounds__(256)
void gi_gemm_kernel(const __bf16* __restrict__ X, const __bf16* __restrict__ W,
                    const float* __restrict__ bif, const float* __restrict__ bib,
                    float* __restrict__ G) {
    const int lane = threadIdx.x & 31, wave = threadIdx.x >> 5;
    const int l16 = lane & 15, lh = lane >> 4;
    const int gw = blockIdx.x * 8 + wave;          // 49152 waves = 2048 mtiles * 24 ngroups
    const int mtile = gw / 24, ngroup = gw % 24;
    const int dir = ngroup / 12;                   // ngroups 0..11 fwd, 12..23 bwd
    const int gbase = (ngroup % 12) * 64;          // column base within this dir's 768
    const float* bi = dir ? bib : bif;

    v8f c[4];
#pragma unroll
    for (int i = 0; i < 4; ++i) {
        float bv = bi[gbase + i * 16 + l16];       // per-column bias, broadcast over rows
#pragma unroll
        for (int v = 0; v < 8; ++v) c[i][v] = bv;
    }

    const __bf16* xrow = X + (size_t)(mtile * 16 + l16) * 256;
    const __bf16* wr[4];
#pragma unroll
    for (int i = 0; i < 4; ++i)
        wr[i] = W + (size_t)(dir * GN + gbase + i * 16 + l16) * 256;

#pragma unroll
    for (int kc = 0; kc < 8; ++kc) {
        Frag a;
        const int ka = kc * 32 + lh * 16;          // A: per-lane contiguous 16 K at lh*16
        a.h[0] = *(const bf16x8*)(xrow + ka);
        a.h[1] = *(const bf16x8*)(xrow + ka + 8);
        const int kb = kc * 32 + lh * 16;          // B: per-lane contiguous 16 K at lh*16
#pragma unroll
        for (int i = 0; i < 4; ++i) {
            Frag b;
            b.h[0] = *(const bf16x8*)(wr[i] + kb);
            b.h[1] = *(const bf16x8*)(wr[i] + kb + 8);
            c[i] = wmma_bf16(a, b, c[i]);
        }
    }

#pragma unroll
    for (int i = 0; i < 4; ++i) {
        const int col = ngroup * 64 + i * 16 + l16;
#pragma unroll
        for (int v = 0; v < 8; ++v) {
            const int m = mtile * 16 + lh * 8 + v; // C/D: row = lh*8 + vgpr
            G[(size_t)m * NC + col] = c[i][v];
        }
    }
}

// ---------------------------------------------------------------------------
// Recurrent scan. grid = 8 blocks: blockIdx = dir*4 + batch_group(16 rows).
// 512 threads = 16 waves; wave w owns hidden columns [16w,16w+16) for ALL
// three gates (Wh row blocks w, w+16, w+32) -> gate math stays in registers.
// Wh B-fragments are loop-invariant -> preloaded into 192 VGPRs per wave,
// so the 512-step loop issues ZERO weight traffic. h is double-buffered in
// LDS (bf16, 2 x 8 KB) as the shared WMMA A operand: one barrier per step.
// Next step's gi cachelines are prefetched during the gate phase.
// ---------------------------------------------------------------------------
__global__ __launch_bounds__(512)
void gru_scan_kernel(const float* __restrict__ G,
                     const __bf16* __restrict__ Whf, const __bf16* __restrict__ Whb,
                     const float* __restrict__ bhf, const float* __restrict__ bhb,
                     const float* __restrict__ h0f, const float* __restrict__ h0b,
                     float* __restrict__ out) {
    __shared__ __bf16 hl[2][16 * 256];             // double-buffered h (2 x 8 KB)

    const int lane = threadIdx.x & 31, wave = threadIdx.x >> 5;
    const int l16 = lane & 15, lh = lane >> 4;
    const int dir = blockIdx.x >> 2, bg = blockIdx.x & 3;

    const __bf16* Wh = dir ? Whb : Whf;
    const float*  bh = dir ? bhb : bhf;
    const float*  h0 = dir ? h0b : h0f;

    const int j = wave * 16 + l16;                 // hidden column 0..255 (this lane)
    const float br_b = bh[0 * Hn + j];
    const float bz_b = bh[1 * Hn + j];
    const float bn_b = bh[2 * Hn + j];

    // ---- preload all Wh B-fragments for this wave's column (loop-invariant) ----
    const __bf16* wr_r = Wh + (size_t)(0 * Hn + j) * 256;
    const __bf16* wr_z = Wh + (size_t)(1 * Hn + j) * 256;
    const __bf16* wr_n = Wh + (size_t)(2 * Hn + j) * 256;
    Frag br[8], bz[8], bn[8];                      // 3 x 8 x 8 = 192 VGPRs
#pragma unroll
    for (int kc = 0; kc < 8; ++kc) {
        const int kb = kc * 32 + lh * 16;          // B: per-lane contiguous 16 K at lh*16
        br[kc].h[0] = *(const bf16x8*)(wr_r + kb);
        br[kc].h[1] = *(const bf16x8*)(wr_r + kb + 8);
        bz[kc].h[0] = *(const bf16x8*)(wr_z + kb);
        bz[kc].h[1] = *(const bf16x8*)(wr_z + kb + 8);
        bn[kc].h[0] = *(const bf16x8*)(wr_n + kb);
        bn[kc].h[1] = *(const bf16x8*)(wr_n + kb + 8);
    }

    // ---- init h into buffer 0 (C-fragment layout: row = lh*8 + v) ----
#pragma unroll
    for (int v = 0; v < 8; ++v) {
        const int m = lh * 8 + v;
        hl[0][m * 256 + j] = (__bf16)h0[(size_t)(bg * 16 + m) * Hn + j];
    }
    __syncthreads();

    int cur = 0;
    for (int s = 0; s < Tn; ++s) {
        const int t = dir ? (Tn - 1 - s) : s;

        v8f cr, cz, cn;
#pragma unroll
        for (int v = 0; v < 8; ++v) { cr[v] = br_b; cz[v] = bz_b; cn[v] = bn_b; }

        const __bf16* hrd = &hl[cur][0];
#pragma unroll
        for (int kc = 0; kc < 8; ++kc) {
            Frag a;
            const int ka = kc * 32 + lh * 16;
            a.h[0] = *(const bf16x8*)(hrd + l16 * 256 + ka);
            a.h[1] = *(const bf16x8*)(hrd + l16 * 256 + ka + 8);
            cr = wmma_bf16(a, br[kc], cr);
            cz = wmma_bf16(a, bz[kc], cz);
            cn = wmma_bf16(a, bn[kc], cn);
        }

        // gates + output; h_new goes to the other LDS buffer (no extra barrier)
        __bf16* hwr = &hl[cur ^ 1][0];
#pragma unroll
        for (int v = 0; v < 8; ++v) {
            const int m = lh * 8 + v;
            const int b = bg * 16 + m;
            const size_t gi = ((size_t)b * Tn + t) * NC + (size_t)dir * GN + j;
            const float hp = (float)hrd[m * 256 + j];
            const float r  = sigmoid_f(G[gi]           + cr[v]);
            const float z  = sigmoid_f(G[gi + Hn]      + cz[v]);
            const float nn = tanh_f   (G[gi + 2 * Hn]  + r * cn[v]);
            const float hv = tanh_f((1.0f - z) * nn + z * hp);
            hwr[m * 256 + j] = (__bf16)hv;
            out[((size_t)b * Tn + t) * (2 * Hn) + dir * Hn + j] = hv;
        }

        // prefetch next step's gi cachelines (hidden behind next K-loop)
        if (s + 1 < Tn) {
            const int tnx = dir ? (t - 1) : (t + 1);
#pragma unroll
            for (int v = 0; v < 8; ++v) {
                const int b = bg * 16 + lh * 8 + v;
                const float* p = G + ((size_t)b * Tn + tnx) * NC + (size_t)dir * GN + j;
                __builtin_prefetch(p,          0, 3);
                __builtin_prefetch(p + Hn,     0, 3);
                __builtin_prefetch(p + 2 * Hn, 0, 3);
            }
        }

        __syncthreads();                            // hl[cur^1] complete for next step
        cur ^= 1;
    }
}

// ---------------------------------------------------------------------------
// Host launcher
// inputs: 0 x[B,T,I] 1 h0_fwd 2 h0_bwd 3 W_ih_f 4 W_hh_f 5 b_ih_f 6 b_hh_f
//         7 W_ih_b 8 W_hh_b 9 b_ih_b 10 b_hh_b 11 lengths (full, unused)
// ---------------------------------------------------------------------------
extern "C" void kernel_launch(void* const* d_in, const int* in_sizes, int n_in,
                              void* d_out, int out_size, void* d_ws, size_t ws_size,
                              hipStream_t stream) {
    const float* x    = (const float*)d_in[0];
    const float* h0f  = (const float*)d_in[1];
    const float* h0b  = (const float*)d_in[2];
    const float* Wif  = (const float*)d_in[3];
    const float* Whf  = (const float*)d_in[4];
    const float* bif  = (const float*)d_in[5];
    const float* bhf  = (const float*)d_in[6];
    const float* Wib  = (const float*)d_in[7];
    const float* Whb  = (const float*)d_in[8];
    const float* bib  = (const float*)d_in[9];
    const float* bhb  = (const float*)d_in[10];
    float* out = (float*)d_out;

    // workspace layout
    char* ws = (char*)d_ws;
    const size_t xN  = (size_t)Bn * Tn * In;          // 8388608
    const size_t wN  = (size_t)GN * In;               // 196608 per weight matrix
    __bf16* xb  = (__bf16*)ws;                        // 16 MB
    __bf16* wic = (__bf16*)(ws + 16777216);           // [Wi_f ; Wi_b] 768 KB
    __bf16* whf = (__bf16*)(ws + 16777216 + 786432);  // 384 KB
    __bf16* whb = (__bf16*)(ws + 16777216 + 786432 + 393216);
    float*  gic = (float*) (ws + 18350080);           // [32768,1536] fp32, 192 MB

    // 1) convert to bf16
    cvt_bf16_kernel<<<(int)((xN + 255) / 256), 256, 0, stream>>>(x, xb, (int)xN);
    cvt_bf16_kernel<<<(int)((wN + 255) / 256), 256, 0, stream>>>(Wif, wic, (int)wN);
    cvt_bf16_kernel<<<(int)((wN + 255) / 256), 256, 0, stream>>>(Wib, wic + wN, (int)wN);
    cvt_bf16_kernel<<<(int)((wN + 255) / 256), 256, 0, stream>>>(Whf, whf, (int)wN);
    cvt_bf16_kernel<<<(int)((wN + 255) / 256), 256, 0, stream>>>(Whb, whb, (int)wN);

    // 2) input projections for both directions: 6144 blocks x 8 waves
    gi_gemm_kernel<<<6144, 256, 0, stream>>>(xb, wic, bif, bib, gic);

    // 3) recurrent scan: 8 persistent workgroups (2 dirs x 4 batch groups)
    gru_scan_kernel<<<8, 512, 0, stream>>>(gic, whf, whb, bhf, bhb, h0f, h0b, out);
}
